// FlashCrossTransformerLayer_20564303413374
// MI455X (gfx1250) — matrix operand order repaired and checked by reading, not compiled, not run on hardware
//
#include <hip/hip_runtime.h>
#include <hip/hip_bf16.h>

typedef __attribute__((ext_vector_type(16))) __bf16          v16bf;
typedef __attribute__((ext_vector_type(16))) unsigned short  v16u;
typedef __attribute__((ext_vector_type(8)))  unsigned short  v8u;
typedef __attribute__((ext_vector_type(8)))  float           v8f;

__device__ __forceinline__ unsigned short f2bf(float f) {
    unsigned int u = __builtin_bit_cast(unsigned int, f);
    unsigned int r = u + 0x7FFFu + ((u >> 16) & 1u);   // round-to-nearest-even
    return (unsigned short)(r >> 16);
}

// ---- CDNA5 async global->LDS copy (ASYNCcnt-tracked), inline asm ----------
// LDS operand = low 32 bits of the generic pointer (ISA 10.2: LDS aperture
// address is addr[31:0]).  GV addressing mode (64-bit vaddr, saddr=off).
__device__ __forceinline__ void async_load_b128(const unsigned short* gptr, void* lptr) {
    asm volatile("global_load_async_to_lds_b128 %0, %1, off"
                 :: "v"((unsigned)reinterpret_cast<unsigned long long>(lptr)),
                    "v"(gptr)
                 : "memory");
}
__device__ __forceinline__ void wait_async0() {
    asm volatile("s_wait_asynccnt 0x0" ::: "memory");
}

// ---- WMMA fragment loaders (CDNA5 ISA 7.12.2 layouts, wave32) -------------
// A: 16x32 bf16.  lane m = lane&15 (M row), th = lane>>4 selects K phase.
// Per lane: two contiguous 8-element (16B) chunks -> 2x ds_load_b128.
__device__ __forceinline__ v16bf load_a_frag(const unsigned short* base, int stride, int lane) {
    const int th = lane >> 4, m = lane & 15;
    v16u u;
#pragma unroll
    for (int v = 0; v < 8; ++v) {
        int kb = ((v < 4) ? (v * 2) : (16 + (v - 4) * 2)) + th * 8;
        u[2 * v]     = base[m * stride + kb];
        u[2 * v + 1] = base[m * stride + kb + 1];
    }
    return __builtin_bit_cast(v16bf, u);
}

// B (from row-major N×K source): Bmat[k][n] = base[n*stride + k].
// lanes 0-15: N=lane, K=0..7,16..23 ; lanes 16-31: N=lane-16, K=8..15,24..31.
// Per lane: two contiguous 8-element (16B) chunks.
__device__ __forceinline__ v16bf load_bT_frag(const unsigned short* base, int stride, int lane) {
    const int th = lane >> 4, n = lane & 15;
    v16u u;
#pragma unroll
    for (int v = 0; v < 8; ++v) {
        int k = ((v < 4) ? (v * 2) : (16 + (v - 4) * 2)) + th * 8;
        u[2 * v]     = base[n * stride + k];
        u[2 * v + 1] = base[n * stride + k + 1];
    }
    return __builtin_bit_cast(v16bf, u);
}

__device__ __forceinline__ float rowmax16(float v) {
#pragma unroll
    for (int m = 8; m >= 1; m >>= 1) v = fmaxf(v, __shfl_xor(v, m, 32));
    return v;
}
__device__ __forceinline__ float rowsum16(float v) {
#pragma unroll
    for (int m = 8; m >= 1; m >>= 1) v += __shfl_xor(v, m, 32);
    return v;
}

// ---- fp32 -> bf16 conversion ---------------------------------------------
__global__ void cvt_bf16(const float* __restrict__ src, unsigned short* __restrict__ dst, int n) {
    int i = blockIdx.x * blockDim.x + threadIdx.x;
    if (i < n) dst[i] = f2bf(src[i]);
}

// ---- Tiled WMMA GEMM: out[M,N] = A[M,K] (bf16) @ W[N,K]^T (bf16) ----------
// block: 256 threads = 8 waves, 64(M)x128(N) tile; wave -> 16x64 subtile.
// Double-buffered LDS fed by global_load_async_to_lds_b128: wait ASYNCcnt for
// the current tile, issue next tile async, compute WMMAs (one barrier/iter).
// SPLIT_KV: K half (cols<512) -> outp [rows x 512]; V half written transposed
// to vt as VT[b, d, s] for the flash P·V stage.
template <int OUT_BF16, int RELU, int HAS_BIAS, int SPLIT_KV>
__global__ __launch_bounds__(256) void gemm_bf16_kernel(
    const unsigned short* __restrict__ A, const unsigned short* __restrict__ W,
    const float* __restrict__ bias, void* __restrict__ outp,
    unsigned short* __restrict__ vt,
    int M, int N, int K)
{
    __shared__ unsigned short As[2][64][40];    // [buf][m][k], padded
    __shared__ unsigned short Ws[2][128][40];   // [buf][n][k], padded row-major
    const int tid = threadIdx.x, lane = tid & 31, wave = tid >> 5;
    const int bm = blockIdx.x * 64, bn = blockIdx.y * 128;
    const int wm = (wave & 3) * 16, wn = (wave >> 2) * 64;
    const int ar = tid >> 2, ac = (tid & 3) * 8;     // A tile: 64 x 32
    const int wr = tid >> 1, wc = (tid & 1) * 16;    // W tile: 128 x 32

    auto issue = [&](int k0, int buf) {
        async_load_b128(A + (size_t)(bm + ar) * K + k0 + ac,     &As[buf][ar][ac]);
        async_load_b128(W + (size_t)(bn + wr) * K + k0 + wc,     &Ws[buf][wr][wc]);
        async_load_b128(W + (size_t)(bn + wr) * K + k0 + wc + 8, &Ws[buf][wr][wc + 8]);
    };
    issue(0, 0);

    v8f acc0 = {}, acc1 = {}, acc2 = {}, acc3 = {};
    for (int k0 = 0; k0 < K; k0 += 32) {
        const int buf = (k0 >> 5) & 1;
        wait_async0();        // this wave's portion of tile(buf) landed in LDS
        __syncthreads();      // -> all waves' portions landed; prev reads done
        if (k0 + 32 < K) issue(k0 + 32, buf ^ 1);

        v16bf af = load_a_frag(&As[buf][wm][0], 40, lane);
        acc0 = __builtin_amdgcn_wmma_f32_16x16x32_bf16(false, af, false, load_bT_frag(&Ws[buf][wn][0],      40, lane), (short)0, acc0, false, false);
        acc1 = __builtin_amdgcn_wmma_f32_16x16x32_bf16(false, af, false, load_bT_frag(&Ws[buf][wn + 16][0], 40, lane), (short)0, acc1, false, false);
        acc2 = __builtin_amdgcn_wmma_f32_16x16x32_bf16(false, af, false, load_bT_frag(&Ws[buf][wn + 32][0], 40, lane), (short)0, acc2, false, false);
        acc3 = __builtin_amdgcn_wmma_f32_16x16x32_bf16(false, af, false, load_bT_frag(&Ws[buf][wn + 48][0], 40, lane), (short)0, acc3, false, false);
    }

    const int th = lane >> 4, tl = lane & 15;
#pragma unroll
    for (int nt = 0; nt < 4; ++nt) {
        v8f acc = (nt == 0) ? acc0 : (nt == 1) ? acc1 : (nt == 2) ? acc2 : acc3;
#pragma unroll
        for (int r = 0; r < 8; ++r) {
            int row = bm + wm + r + 8 * th;
            int c = bn + wn + nt * 16 + tl;
            float v = acc[r];
            if (HAS_BIAS) v += bias[c];
            if (RELU)     v = fmaxf(v, 0.f);
            if (SPLIT_KV) {
                if (c < 512) {
                    ((unsigned short*)outp)[(size_t)row * 512 + c] = f2bf(v);
                } else {
                    int b = row >> 11, s = row & 2047;   // S = 2048
                    vt[((size_t)b * 512 + (c - 512)) * 2048 + s] = f2bf(v);
                }
            } else if (OUT_BF16) {
                ((unsigned short*)outp)[(size_t)row * N + c] = f2bf(v);
            } else {
                ((float*)outp)[(size_t)row * N + c] = v;
            }
        }
    }
}

// ---- Flash cross-attention ------------------------------------------------
// grid: B*H*(V/64) blocks of 128 threads (4 waves).  Each wave owns 16 q rows.
// Q: [B*V, 512] bf16 (head h at col h*64).  Kb: [B*S, 512] bf16.
// VTb: [B, 512, S] bf16 (d-major -> contiguous-s fragment loads).
// K/V tiles double-buffered via async global->LDS.  O: [B*V, 512] bf16.
__global__ __launch_bounds__(128) void flash_attn_kernel(
    const unsigned short* __restrict__ Q, const unsigned short* __restrict__ Kb,
    const unsigned short* __restrict__ VTb, const unsigned char* __restrict__ mask,
    unsigned short* __restrict__ O)
{
    const int V = 1024, S = 2048, D = 512, HD = 64;
    const int tid = threadIdx.x, lane = tid & 31, wave = tid >> 5;
    const int th = lane >> 4, tl = lane & 15;
    const int qt = blockIdx.x & 15, h = (blockIdx.x >> 4) & 7, b = blockIdx.x >> 7;
    const int q0 = qt * 64;

    __shared__ unsigned short Kt[2][64][72];    // [buf][s][d]
    __shared__ unsigned short VtT[2][64][72];   // [buf][d][s]
    __shared__ unsigned short Pt[4][16][72];    // per-wave P scratch (C->A relayout)

    const unsigned short* qrow = Q + (size_t)(b * V + q0 + wave * 16) * D + h * HD;
    v16bf qa0 = load_a_frag(qrow, D, lane);
    v16bf qa1 = load_a_frag(qrow + 32, D, lane);

    const int ls = tid >> 1, lc = (tid & 1) * 32;
    auto issueKV = [&](int s0, int buf) {
#pragma unroll
        for (int j = 0; j < 4; ++j) {
            async_load_b128(Kb  + (size_t)(b * S + s0 + ls) * 512 + h * HD + lc + j * 8,
                            &Kt[buf][ls][lc + j * 8]);
            async_load_b128(VTb + (size_t)(b * 512 + h * HD + ls) * 2048 + s0 + lc + j * 8,
                            &VtT[buf][ls][lc + j * 8]);
        }
    };
    issueKV(0, 0);

    float mrow[8], lrow[8];
    v8f o0 = {}, o1 = {}, o2 = {}, o3 = {};
#pragma unroll
    for (int r = 0; r < 8; ++r) { mrow[r] = -1e30f; lrow[r] = 0.f; }

    for (int s0 = 0; s0 < S; s0 += 64) {
        const int buf = (s0 >> 6) & 1;
        wait_async0();
        __syncthreads();
        if (s0 + 64 < S) issueKV(s0 + 64, buf ^ 1);

        v8f sc[4];
#pragma unroll
        for (int nt = 0; nt < 4; ++nt) {
            v8f z = {};
            v16bf bk0 = load_bT_frag(&Kt[buf][nt * 16][0], 72, lane);
            v16bf bk1 = load_bT_frag(&Kt[buf][nt * 16][32], 72, lane);
            z = __builtin_amdgcn_wmma_f32_16x16x32_bf16(false, qa0, false, bk0, (short)0, z, false, false);
            z = __builtin_amdgcn_wmma_f32_16x16x32_bf16(false, qa1, false, bk1, (short)0, z, false, false);
            float mbias = mask[(size_t)b * S + s0 + nt * 16 + tl] ? -1e30f : 0.f;
#pragma unroll
            for (int r = 0; r < 8; ++r) z[r] = z[r] * 0.125f + mbias;   // 1/sqrt(64)
            sc[nt] = z;
        }

        float alpha[8];
#pragma unroll
        for (int r = 0; r < 8; ++r) {
            float t = fmaxf(fmaxf(sc[0][r], sc[1][r]), fmaxf(sc[2][r], sc[3][r]));
            t = rowmax16(t);
            float mn = fmaxf(mrow[r], t);
            alpha[r] = __expf(mrow[r] - mn);
            mrow[r] = mn;
        }
#pragma unroll
        for (int nt = 0; nt < 4; ++nt)
#pragma unroll
            for (int r = 0; r < 8; ++r) sc[nt][r] = __expf(sc[nt][r] - mrow[r]);
#pragma unroll
        for (int r = 0; r < 8; ++r) {
            float rs = sc[0][r] + sc[1][r] + sc[2][r] + sc[3][r];
            rs = rowsum16(rs);
            lrow[r] = lrow[r] * alpha[r] + rs;
            o0[r] *= alpha[r]; o1[r] *= alpha[r]; o2[r] *= alpha[r]; o3[r] *= alpha[r];
        }

        // C-layout -> A-layout via wave-private LDS
#pragma unroll
        for (int nt = 0; nt < 4; ++nt)
#pragma unroll
            for (int r = 0; r < 8; ++r)
                Pt[wave][r + 8 * th][nt * 16 + tl] = f2bf(sc[nt][r]);

#pragma unroll
        for (int f = 0; f < 2; ++f) {
            v16bf pa = load_a_frag(&Pt[wave][0][f * 32], 72, lane);
            o0 = __builtin_amdgcn_wmma_f32_16x16x32_bf16(false, pa, false, load_bT_frag(&VtT[buf][0][f * 32],  72, lane), (short)0, o0, false, false);
            o1 = __builtin_amdgcn_wmma_f32_16x16x32_bf16(false, pa, false, load_bT_frag(&VtT[buf][16][f * 32], 72, lane), (short)0, o1, false, false);
            o2 = __builtin_amdgcn_wmma_f32_16x16x32_bf16(false, pa, false, load_bT_frag(&VtT[buf][32][f * 32], 72, lane), (short)0, o2, false, false);
            o3 = __builtin_amdgcn_wmma_f32_16x16x32_bf16(false, pa, false, load_bT_frag(&VtT[buf][48][f * 32], 72, lane), (short)0, o3, false, false);
        }
    }

#pragma unroll
    for (int r = 0; r < 8; ++r) {
        float inv = 1.f / lrow[r];
        size_t base = (size_t)(b * V + q0 + wave * 16 + r + 8 * th) * D + h * HD;
        O[base + tl]      = f2bf(o0[r] * inv);
        O[base + 16 + tl] = f2bf(o1[r] * inv);
        O[base + 32 + tl] = f2bf(o2[r] * inv);
        O[base + 48 + tl] = f2bf(o3[r] * inv);
    }
}

// ---- residual + LayerNorm (D=512, one row per block) ----------------------
template <int WRITE_BF>
__global__ __launch_bounds__(256) void ln_kernel(
    const float* __restrict__ a, const float* __restrict__ bres,
    const float* __restrict__ g, const float* __restrict__ be,
    float* __restrict__ outf, unsigned short* __restrict__ outb)
{
    const int D = 512;
    __shared__ float red[256];
    const int row = blockIdx.x, t = threadIdx.x;
    const size_t base = (size_t)row * D;
    float v0 = a[base + t] + bres[base + t];
    float v1 = a[base + 256 + t] + bres[base + 256 + t];
    red[t] = v0 + v1;
    __syncthreads();
#pragma unroll
    for (int s = 128; s > 0; s >>= 1) { if (t < s) red[t] += red[t + s]; __syncthreads(); }
    float mu = red[0] * (1.f / 512.f);
    __syncthreads();
    float d0 = v0 - mu, d1 = v1 - mu;
    red[t] = d0 * d0 + d1 * d1;
    __syncthreads();
#pragma unroll
    for (int s = 128; s > 0; s >>= 1) { if (t < s) red[t] += red[t + s]; __syncthreads(); }
    float rstd = rsqrtf(red[0] * (1.f / 512.f) + 1e-5f);
    float y0 = d0 * rstd * g[t] + be[t];
    float y1 = d1 * rstd * g[t + 256] + be[t + 256];
    outf[base + t] = y0;
    outf[base + 256 + t] = y1;
    if (WRITE_BF) { outb[base + t] = f2bf(y0); outb[base + 256 + t] = f2bf(y1); }
}

// ---------------------------------------------------------------------------
extern "C" void kernel_launch(void* const* d_in, const int* in_sizes, int n_in,
                              void* d_out, int out_size, void* d_ws, size_t ws_size,
                              hipStream_t stream)
{
    const float* tgt  = (const float*)d_in[0];
    const float* memv = (const float*)d_in[1];
    const unsigned char* mask = (const unsigned char*)d_in[2];
    const float* Wq  = (const float*)d_in[3];
    const float* Wkv = (const float*)d_in[4];
    const float* Wo  = (const float*)d_in[5];
    const float* W1  = (const float*)d_in[6];
    const float* b1  = (const float*)d_in[7];
    const float* W2  = (const float*)d_in[8];
    const float* b2  = (const float*)d_in[9];
    const float* g1  = (const float*)d_in[10];
    const float* be1 = (const float*)d_in[11];
    const float* g2  = (const float*)d_in[12];
    const float* be2 = (const float*)d_in[13];

    const int B = 4, V = 1024, S = 2048, D = 512, FF = 2048;
    const int BV = B * V, BS = B * S;

    char* w = (char*)d_ws;
    auto alloc = [&](size_t bytes) -> char* {
        char* p = w; w += (bytes + 255) & ~(size_t)255; return p;
    };
    unsigned short* tgt_bf = (unsigned short*)alloc((size_t)BV * D * 2);
    unsigned short* mem_bf = (unsigned short*)alloc((size_t)BS * D * 2);
    unsigned short* Wq_bf  = (unsigned short*)alloc((size_t)D * D * 2);
    unsigned short* Wkv_bf = (unsigned short*)alloc((size_t)2 * D * D * 2);
    unsigned short* Wo_bf  = (unsigned short*)alloc((size_t)D * D * 2);
    unsigned short* W1_bf  = (unsigned short*)alloc((size_t)FF * D * 2);
    unsigned short* W2_bf  = (unsigned short*)alloc((size_t)D * FF * 2);
    unsigned short* Q_bf   = (unsigned short*)alloc((size_t)BV * D * 2);
    unsigned short* K_bf   = (unsigned short*)alloc((size_t)BS * D * 2);
    unsigned short* VT_bf  = (unsigned short*)alloc((size_t)B * D * S * 2);
    unsigned short* A_bf   = (unsigned short*)alloc((size_t)BV * D * 2);
    float*          AO_f   = (float*)alloc((size_t)BV * D * 4);
    float*          X_f    = (float*)alloc((size_t)BV * D * 4);
    unsigned short* X_bf   = (unsigned short*)alloc((size_t)BV * D * 2);
    unsigned short* F1_bf  = (unsigned short*)alloc((size_t)BV * FF * 2);
    float*          F2_f   = (float*)alloc((size_t)BV * D * 4);

    auto cvt = [&](const float* s, unsigned short* d, int n) {
        cvt_bf16<<<(n + 255) / 256, 256, 0, stream>>>(s, d, n);
    };
    cvt(tgt,  tgt_bf, BV * D);
    cvt(memv, mem_bf, BS * D);
    cvt(Wq,   Wq_bf,  D * D);
    cvt(Wkv,  Wkv_bf, 2 * D * D);
    cvt(Wo,   Wo_bf,  D * D);
    cvt(W1,   W1_bf,  FF * D);
    cvt(W2,   W2_bf,  D * FF);

    // Q = tgt @ Wq^T   (bf16 out)
    gemm_bf16_kernel<1, 0, 0, 0><<<dim3(BV / 64, D / 128), 256, 0, stream>>>(
        tgt_bf, Wq_bf, nullptr, Q_bf, nullptr, BV, D, D);
    // KV = memory @ Wkv^T  -> K_bf [BS x 512] + V transposed into VT_bf [B,512,S]
    gemm_bf16_kernel<1, 0, 0, 1><<<dim3(BS / 64, (2 * D) / 128), 256, 0, stream>>>(
        mem_bf, Wkv_bf, nullptr, K_bf, VT_bf, BS, 2 * D, D);
    // fused masked flash attention -> bf16 attn_out
    flash_attn_kernel<<<B * 8 * (V / 64), 128, 0, stream>>>(Q_bf, K_bf, VT_bf, mask, A_bf);
    // attn_out @ Wo^T  (f32 out)
    gemm_bf16_kernel<0, 0, 0, 0><<<dim3(BV / 64, D / 128), 256, 0, stream>>>(
        A_bf, Wo_bf, nullptr, AO_f, nullptr, BV, D, D);
    // x = LN(tgt + attn_proj); also emit bf16 copy for FFN
    ln_kernel<1><<<BV, 256, 0, stream>>>(tgt, AO_f, g1, be1, X_f, X_bf);
    // ff1 = relu(x @ W1^T + b1)  (bf16 out)
    gemm_bf16_kernel<1, 1, 1, 0><<<dim3(BV / 64, FF / 128), 256, 0, stream>>>(
        X_bf, W1_bf, b1, F1_bf, nullptr, BV, FF, D);
    // ff2 = ff1 @ W2^T + b2  (f32 out)
    gemm_bf16_kernel<0, 0, 1, 0><<<dim3(BV / 64, D / 128), 256, 0, stream>>>(
        F1_bf, W2_bf, b2, F2_f, nullptr, BV, D, FF);
    // out = LN(x + ff2)
    ln_kernel<0><<<BV, 256, 0, stream>>>(X_f, F2_f, g2, be2, (float*)d_out, nullptr);
}
